// RestrictedSelfAttention_9002251453069
// MI455X (gfx1250) — matrix operand order repaired and checked by reading, not compiled, not run on hardware
//
#include <hip/hip_runtime.h>
#include <hip/hip_bf16.h>

// Problem constants (reference: BS, N, D, R, NH = 2, 8192, 256, 32, 8; DH=32)
#define BSZ    2
#define SEQ    8192
#define DM     256
#define RNB    32
#define NHD    8
#define MTOT   (BSZ * SEQ)          // 16384 rows

typedef __attribute__((ext_vector_type(16))) __bf16        v16bf;
typedef __attribute__((ext_vector_type(8)))  __bf16        v8bf;
typedef __attribute__((ext_vector_type(8)))  float         v8f;
typedef __attribute__((ext_vector_type(8)))  unsigned short u16x8;

__device__ __forceinline__ unsigned short f2bf(float f) {
  unsigned u = __builtin_bit_cast(unsigned, f);
  u += 0x7FFFu + ((u >> 16) & 1u);            // round-to-nearest-even
  return (unsigned short)(u >> 16);
}
__device__ __forceinline__ float bf2f(unsigned short h) {
  return __builtin_bit_cast(float, (unsigned)h << 16);
}

// ---------------------------------------------------------------- convert
__global__ __launch_bounds__(256) void cvt_f32_to_bf16(
    const float* __restrict__ s, unsigned short* __restrict__ d, int n) {
  int i = (blockIdx.x * 256 + threadIdx.x) * 4;
  if (i + 3 < n) {
    float4 v = *(const float4*)(s + i);
    ushort4 o = make_ushort4(f2bf(v.x), f2bf(v.y), f2bf(v.z), f2bf(v.w));
    *(ushort4*)(d + i) = o;
  }
}

// ---------------------------------------------------------------- GEMM
// C(M x 256) = A(M x 256, bf16) @ W^T + bias, W is 256x256 row-major (n,k),
// so B[k,n] = W[n,k]: WMMA B lane n reads W row n contiguously.
// Wave computes a 32(M) x 64(N) tile: 2 A fragments x 4 B fragments = 8 WMMAs
// per k-step, with explicit double-buffering so fragment loads for step k+1
// overlap the 8 WMMAs of step k (one loadcnt wait per step instead of per WMMA).
template <bool OUT_BF16>
__global__ __launch_bounds__(256) void gemm256_bf16_wmma(
    const unsigned short* __restrict__ A,
    const unsigned short* __restrict__ W,
    const float* __restrict__ bias,
    void* __restrict__ outp, int M) {
  const int lane = threadIdx.x & 31;
  const int wave = threadIdx.x >> 5;
  const int m0 = (blockIdx.x * 8 + wave) * 32;
  const int n0 = blockIdx.y * 64;
  if (m0 >= M) return;

  // ISA 16-bit A 16x32 layout: lanes0-15 K={0..7,16..23}, lanes16-31 K={8..15,24..31}
  const int mr  = lane & 15;
  const int khA = (lane >> 4) * 8;
  // ISA 16-bit B 32x16 layout: lanes0-15 K={0..7,16..23}, lanes16-31 K={8..15,24..31}, N=lane&15
  const int ncol = lane & 15;
  const int kbB  = (lane >> 4) * 8;

  const unsigned short* arow[2];
  arow[0] = A + (size_t)(m0 + mr) * DM + khA;
  arow[1] = arow[0] + (size_t)16 * DM;
  const unsigned short* wrow[4];
#pragma unroll
  for (int nt = 0; nt < 4; ++nt)
    wrow[nt] = W + (size_t)(n0 + nt * 16 + ncol) * DM + kbB;

  v8f acc[2][4] = {};
  v16bf aF[2][2];   // [stage][mtile]
  v16bf bF[2][4];   // [stage][ntile]

  auto loadStage = [&](int st, int k0) {
#pragma unroll
    for (int mt = 0; mt < 2; ++mt) {
      v8bf lo = *(const v8bf*)(arow[mt] + k0);
      v8bf hi = *(const v8bf*)(arow[mt] + k0 + 16);
      aF[st][mt] = __builtin_shufflevector(lo, hi, 0, 1, 2, 3, 4, 5, 6, 7, 8, 9,
                                           10, 11, 12, 13, 14, 15);
    }
#pragma unroll
    for (int nt = 0; nt < 4; ++nt) {
      v8bf lo = *(const v8bf*)(wrow[nt] + k0);
      v8bf hi = *(const v8bf*)(wrow[nt] + k0 + 16);
      bF[st][nt] = __builtin_shufflevector(lo, hi, 0, 1, 2, 3, 4, 5, 6, 7, 8, 9,
                                           10, 11, 12, 13, 14, 15);
    }
  };

  loadStage(0, 0);
#pragma unroll
  for (int ks = 0; ks < 8; ++ks) {
    const int cur = ks & 1;
    if (ks < 7) loadStage(cur ^ 1, (ks + 1) * 32);
#pragma unroll
    for (int mt = 0; mt < 2; ++mt)
#pragma unroll
      for (int nt = 0; nt < 4; ++nt)
        acc[mt][nt] = __builtin_amdgcn_wmma_f32_16x16x32_bf16(
            false, aF[cur][mt], false, bF[cur][nt], (short)0, acc[mt][nt],
            false, false);
  }

  // C/D layout: VGPR v -> row (tile_m + (lane>=16 ? 8:0) + v), col = lane&15
#pragma unroll
  for (int mt = 0; mt < 2; ++mt) {
    const int rbase = m0 + mt * 16 + (lane >> 4) * 8;
#pragma unroll
    for (int nt = 0; nt < 4; ++nt) {
      const int col = n0 + nt * 16 + ncol;
      const float bv = bias[col];
#pragma unroll
      for (int v = 0; v < 8; ++v) {
        float val = acc[mt][nt][v] + bv;
        if (OUT_BF16)
          ((unsigned short*)outp)[(size_t)(rbase + v) * DM + col] = f2bf(val);
        else
          ((float*)outp)[(size_t)(rbase + v) * DM + col] = val;
      }
    }
  }
}

// ---------------------------------------------------------------- attention
// One wave per token. Channel c = e*8 + h (head fastest).
// Phase1: lane = neighbor r -> each lane streams its own K row; q broadcast
//         from LDS; 8-chunk accumulates directly into per-head partials.
// Softmax: butterfly over lanes (r) per head.
// Phase2: lane = channel group e; weights broadcast from LDS; V gathered bf16.
__global__ __launch_bounds__(256) void attn_nbhd_kernel(
    const float* __restrict__ Q, const unsigned short* __restrict__ K,
    const unsigned short* __restrict__ V, const float* __restrict__ P,
    const long long* __restrict__ nidx, unsigned short* __restrict__ O) {
  __shared__ float s_q[8][DM];
  __shared__ float s_w[8][RNB * NHD];
  __shared__ int s_idx[8][RNB];

  const int lane = threadIdx.x & 31;
  const int wave = threadIdx.x >> 5;
  const int t = blockIdx.x * 8 + wave;     // global token, 0..16383
  const int b = t >> 13;                   // / SEQ
  const size_t baseKV = (size_t)b * SEQ;

  {  // stage q into LDS (coalesced, 32B/lane)
    const float4* qp = (const float4*)(Q + (size_t)t * DM + lane * 8);
    float4 q0 = qp[0], q1 = qp[1];
    *(float4*)&s_q[wave][lane * 8] = q0;
    *(float4*)&s_q[wave][lane * 8 + 4] = q1;
  }
  const int myj = (int)nidx[(size_t)t * RNB + lane];
  s_idx[wave][lane] = myj;
  __syncthreads();

  // ---- phase 1: scores for neighbor r = lane
  float sc[8] = {0.f, 0.f, 0.f, 0.f, 0.f, 0.f, 0.f, 0.f};
  {
    const unsigned short* krow = K + (baseKV + (size_t)myj) * DM;
    const unsigned short* vrow = V + (baseKV + (size_t)myj) * DM;
    // warm the near cache with this token's V rows for phase 2 (512B each)
    __builtin_prefetch(vrow, 0, 3);
    __builtin_prefetch(vrow + 64, 0, 3);
    __builtin_prefetch(vrow + 128, 0, 3);
    __builtin_prefetch(vrow + 192, 0, 3);
#pragma unroll
    for (int e = 0; e < 32; ++e) {
      u16x8 kv = *(const u16x8*)(krow + e * 8);
      const float4 qa = *(const float4*)&s_q[wave][e * 8];
      const float4 qb = *(const float4*)&s_q[wave][e * 8 + 4];
      sc[0] += qa.x * bf2f(kv[0]);
      sc[1] += qa.y * bf2f(kv[1]);
      sc[2] += qa.z * bf2f(kv[2]);
      sc[3] += qa.w * bf2f(kv[3]);
      sc[4] += qb.x * bf2f(kv[4]);
      sc[5] += qb.y * bf2f(kv[5]);
      sc[6] += qb.z * bf2f(kv[6]);
      sc[7] += qb.w * bf2f(kv[7]);
    }
  }
  {  // scale + positional bias P[b,n,r=lane,h]
    const float4* pp = (const float4*)(P + (size_t)t * (RNB * NHD) + lane * NHD);
    float4 p0 = pp[0], p1 = pp[1];
    const float scl = 0.17677669529663689f;  // 1/sqrt(32)
    sc[0] = sc[0] * scl + p0.x;
    sc[1] = sc[1] * scl + p0.y;
    sc[2] = sc[2] * scl + p0.z;
    sc[3] = sc[3] * scl + p0.w;
    sc[4] = sc[4] * scl + p1.x;
    sc[5] = sc[5] * scl + p1.y;
    sc[6] = sc[6] * scl + p1.z;
    sc[7] = sc[7] * scl + p1.w;
  }
  // ---- softmax over r (= lanes) per head
  float mx[8], ex[8], sm[8];
#pragma unroll
  for (int h = 0; h < 8; ++h) mx[h] = sc[h];
#pragma unroll
  for (int off = 16; off > 0; off >>= 1)
#pragma unroll
    for (int h = 0; h < 8; ++h) mx[h] = fmaxf(mx[h], __shfl_xor(mx[h], off, 32));
#pragma unroll
  for (int h = 0; h < 8; ++h) {
    ex[h] = __expf(sc[h] - mx[h]);
    sm[h] = ex[h];
  }
#pragma unroll
  for (int off = 16; off > 0; off >>= 1)
#pragma unroll
    for (int h = 0; h < 8; ++h) sm[h] += __shfl_xor(sm[h], off, 32);
#pragma unroll
  for (int h = 0; h < 8; ++h) s_w[wave][lane * NHD + h] = ex[h] / sm[h];
  __syncthreads();

  // ---- phase 2: lane = channel group e; O[c] = sum_r w[r][h]*V[j_r][c]
  float acc[8] = {0.f, 0.f, 0.f, 0.f, 0.f, 0.f, 0.f, 0.f};
#pragma unroll 4
  for (int r = 0; r < 32; ++r) {
    const int j = s_idx[wave][r];
    const float4 wa = *(const float4*)&s_w[wave][r * NHD];
    const float4 wb = *(const float4*)&s_w[wave][r * NHD + 4];
    u16x8 vv = *(const u16x8*)(V + (baseKV + (size_t)j) * DM + lane * 8);
    acc[0] += wa.x * bf2f(vv[0]);
    acc[1] += wa.y * bf2f(vv[1]);
    acc[2] += wa.z * bf2f(vv[2]);
    acc[3] += wa.w * bf2f(vv[3]);
    acc[4] += wb.x * bf2f(vv[4]);
    acc[5] += wb.y * bf2f(vv[5]);
    acc[6] += wb.z * bf2f(vv[6]);
    acc[7] += wb.w * bf2f(vv[7]);
  }
  u16x8 o;
#pragma unroll
  for (int h = 0; h < 8; ++h) o[h] = f2bf(acc[h]);
  *(u16x8*)(O + (size_t)t * DM + lane * 8) = o;
}

// ---------------------------------------------------------------- launch
extern "C" void kernel_launch(void* const* d_in, const int* in_sizes, int n_in,
                              void* d_out, int out_size, void* d_ws,
                              size_t ws_size, hipStream_t stream) {
  const float* X      = (const float*)d_in[0];
  const float* P      = (const float*)d_in[1];
  const long long* NI = (const long long*)d_in[2];
  const float* Wq = (const float*)d_in[3];
  const float* bq = (const float*)d_in[4];
  const float* Wk = (const float*)d_in[5];
  const float* bk = (const float*)d_in[6];
  const float* Wv = (const float*)d_in[7];
  const float* bv = (const float*)d_in[8];
  const float* Wo = (const float*)d_in[9];
  const float* bo = (const float*)d_in[10];
  float* out = (float*)d_out;

  // workspace layout (bf16 buffers as ushort)
  const size_t MD = (size_t)MTOT * DM;     // 4,194,304 elements
  const size_t WD = (size_t)DM * DM;       // 65,536 elements
  unsigned short* Xb  = (unsigned short*)d_ws;
  unsigned short* Wqb = Xb + MD;
  unsigned short* Wkb = Wqb + WD;
  unsigned short* Wvb = Wkb + WD;
  unsigned short* Wob = Wvb + WD;
  unsigned short* Kb  = Wob + WD;
  unsigned short* Vb  = Kb + MD;
  unsigned short* Ob  = Vb + MD;
  float* Qf = (float*)(Ob + MD);           // fp32 Q for score accuracy

  // 1) convert X + weights to bf16 (X reused by 3 GEMMs)
  cvt_f32_to_bf16<<<(int)(MD / 1024), 256, 0, stream>>>(X, Xb, (int)MD);
  cvt_f32_to_bf16<<<(int)(WD / 1024), 256, 0, stream>>>(Wq, Wqb, (int)WD);
  cvt_f32_to_bf16<<<(int)(WD / 1024), 256, 0, stream>>>(Wk, Wkb, (int)WD);
  cvt_f32_to_bf16<<<(int)(WD / 1024), 256, 0, stream>>>(Wv, Wvb, (int)WD);
  cvt_f32_to_bf16<<<(int)(WD / 1024), 256, 0, stream>>>(Wo, Wob, (int)WD);

  // 2) Q/K/V projections (WMMA bf16, fp32 accumulate); 32x64 tile per wave
  dim3 gg(MTOT / 256, DM / 64);
  gemm256_bf16_wmma<false><<<gg, 256, 0, stream>>>(Xb, Wqb, bq, Qf, MTOT);
  gemm256_bf16_wmma<true><<<gg, 256, 0, stream>>>(Xb, Wkb, bk, Kb, MTOT);
  gemm256_bf16_wmma<true><<<gg, 256, 0, stream>>>(Xb, Wvb, bv, Vb, MTOT);

  // 3) neighborhood attention (bf16 K/V gathers stay L2-resident)
  attn_nbhd_kernel<<<MTOT / 8, 256, 0, stream>>>(Qf, Kb, Vb, P, NI, Ob);

  // 4) output projection -> fp32 d_out
  gemm256_bf16_wmma<false><<<gg, 256, 0, stream>>>(Ob, Wob, bo, out, MTOT);
}
